// GraphConvNet_61907658605025
// MI455X (gfx1250) — matrix operand order repaired and checked by reading, not compiled, not run on hardware
//
#include <hip/hip_runtime.h>
#include <hip/hip_bf16.h>

typedef __attribute__((ext_vector_type(16))) _Float16 v16h;
typedef __attribute__((ext_vector_type(8)))  _Float16 v8h;
typedef __attribute__((ext_vector_type(8)))  float    v8f;

// ---------------- degree / norm ----------------

__global__ void gcn_degree_kernel(const int* __restrict__ src,
                                  const int* __restrict__ dst,
                                  float* __restrict__ degO,
                                  float* __restrict__ degI, int E) {
    int stride = gridDim.x * blockDim.x;
    for (int e = blockIdx.x * blockDim.x + threadIdx.x; e < E; e += stride) {
        atomicAdd(&degO[src[e]], 1.0f);
        atomicAdd(&degI[dst[e]], 1.0f);
    }
}

__global__ void gcn_norm_kernel(const float* __restrict__ degO,
                                const float* __restrict__ degI,
                                float* __restrict__ normS,
                                float* __restrict__ normD, int n) {
    int i = blockIdx.x * blockDim.x + threadIdx.x;
    if (i < n) {
        normS[i] = rsqrtf(fmaxf(degO[i], 1.0f));
        normD[i] = rsqrtf(fmaxf(degI[i], 1.0f));
    }
}

// ---------------- weight packing into WMMA B-operand register image ----------
// Layout: [kt][nt][lane 0..31][16 halves], where for lane l, half j:
//   K = kt*32 + (j>=8 ? 16 : 0) + (l>=16 ? 8 : 0) + (j&7), N = nt*16 + (l&15)

__global__ void gcn_packw_kernel(const float* __restrict__ W, _Float16* __restrict__ Bp,
                                 int Kreal, int Nreal, int Kpad, int Npad) {
    int total = Kpad * Npad;
    int idx = blockIdx.x * blockDim.x + threadIdx.x;
    if (idx >= total) return;
    int NT = Npad >> 4;
    int tile = idx >> 9;          // 512 halves per (kt,nt) tile
    int r    = idx & 511;
    int lane = r >> 4;
    int j    = r & 15;
    int kt = tile / NT, nt = tile % NT;
    int K = kt * 32 + ((j >> 3) << 4) + ((lane >> 4) << 3) + (j & 7);
    int N = nt * 16 + (lane & 15);
    float v = (K < Kreal && N < Nreal) ? W[K * Nreal + N] : 0.0f;
    Bp[idx] = (_Float16)v;
}

__global__ void gcn_padbias_kernel(const float* __restrict__ b, float* __restrict__ bp,
                                   int Nreal, int Npad) {
    int i = blockIdx.x * blockDim.x + threadIdx.x;
    if (i < Npad) bp[i] = (i < Nreal) ? b[i] : 0.0f;
}

// ---------------- edge scatter-add: agg[dst] += h[src] * norm_src[src] ------

__global__ void gcn_scatter_kernel(const float* __restrict__ h, int ldh,
                                   const int* __restrict__ src,
                                   const int* __restrict__ dst,
                                   const float* __restrict__ normS,
                                   float* __restrict__ agg, int ldagg,
                                   int Kreal, int E) {
    int lane   = threadIdx.x & 31;
    int wave   = (blockIdx.x * blockDim.x + threadIdx.x) >> 5;
    int nwaves = (gridDim.x * blockDim.x) >> 5;
    for (int e = wave; e < E; e += nwaves) {
        int s = src[e];
        int d = dst[e];
        // prefetch next edge's source row (emits global_prefetch_b8)
        int e2 = e + nwaves;
        if (e2 < E) {
            int s2 = src[e2];
            __builtin_prefetch(h + (size_t)s2 * ldh + lane * 4, 0, 0);
        }
        float ns = normS[s];
        const float* hrow = h + (size_t)s * ldh;
        float* arow = agg + (size_t)d * ldagg;
        for (int c = lane; c < Kreal; c += 32)
            atomicAdd(&arow[c], hrow[c] * ns);
    }
}

// ---------------- agg * norm_dst -> f16 (padded cols already zero) ----------

__global__ void gcn_convert_kernel(const float* __restrict__ agg,
                                   const float* __restrict__ normD,
                                   _Float16* __restrict__ h16,
                                   int Kpad, int total) {
    int i = blockIdx.x * blockDim.x + threadIdx.x;
    if (i >= total) return;
    int row = i / Kpad;
    h16[i] = (_Float16)(agg[i] * normD[row]);
}

// ---------------- WMMA GEMM: C = act(A(f16) @ Bpacked + bias) ---------------
// one wave per 16x16 output tile; KT steps of K=32 via v_wmma_f32_16x16x32_f16

__global__ void gcn_gemm_wmma_kernel(const _Float16* __restrict__ A, int lda,
                                     const _Float16* __restrict__ Bp,
                                     const float* __restrict__ bias,
                                     float* __restrict__ C, int ldc,
                                     int numNT, int KT, int act, int totalTiles) {
    int wave = threadIdx.x >> 5;
    int lane = threadIdx.x & 31;
    int tile = blockIdx.x * (blockDim.x >> 5) + wave;
    if (tile >= totalTiles) return;            // uniform per-wave: EXEC stays all-1s
    int mt = tile / numNT;
    int nt = tile % numNT;

    int row  = (mt << 4) + (lane & 15);
    int koff = (lane >> 4) << 3;               // lanes 0-15: K 0..7 / 16..23 ; 16-31: +8
    const _Float16* arow = A + (size_t)row * lda + koff;
    const _Float16* bptr = Bp + ((size_t)nt * 32 + lane) * 16;
    size_t bstride = (size_t)numNT * 512;      // halves per kt step

    v8f acc = {};
    for (int kt = 0; kt < KT; ++kt) {
        v16h a;
        *(v8h*)&a         = *(const v8h*)(arow + kt * 32);
        *(((v8h*)&a) + 1) = *(const v8h*)(arow + kt * 32 + 16);
        v16h b = *(const v16h*)(bptr + (size_t)kt * bstride);
        acc = __builtin_amdgcn_wmma_f32_16x16x32_f16(
            /*neg_a=*/false, a, /*neg_b=*/false, b,
            /*c_mod=*/(short)0, acc, /*reuse_a=*/false, /*reuse_b=*/false);
    }

    int col = (nt << 4) + (lane & 15);
    float bv = bias[col];
    int rbase = (mt << 4) + ((lane >> 4) << 3);  // lanes 16-31 hold M = r+8
#pragma unroll
    for (int r = 0; r < 8; ++r) {
        float v = acc[r] + bv;
        if (act == 0) v = (v > 0.0f) ? v : expm1f(v);          // ELU
        else          v = 1.0f / (1.0f + __expf(-v));          // sigmoid
        C[(size_t)(rbase + r) * ldc + col] = v;
    }
}

// ---------------- host orchestration ----------------

extern "C" void kernel_launch(void* const* d_in, const int* in_sizes, int n_in,
                              void* d_out, int out_size, void* d_ws, size_t ws_size,
                              hipStream_t stream) {
    (void)n_in; (void)out_size; (void)ws_size;
    const int IN_F = 128, H1 = 100, H2 = 200, OUTF = 64;
    const int H1P = 128, H2P = 224;                // padded to multiples of 32
    const int N = in_sizes[0] / IN_F;              // 100000
    const int E = in_sizes[1];                     // 1600000

    const float* x   = (const float*)d_in[0];
    const int*   es  = (const int*)  d_in[1];
    const int*   ed  = (const int*)  d_in[2];
    const float* W1  = (const float*)d_in[3];
    const float* b1  = (const float*)d_in[4];
    const float* W2  = (const float*)d_in[5];
    const float* b2  = (const float*)d_in[6];
    const float* W3  = (const float*)d_in[7];
    const float* b3  = (const float*)d_in[8];
    float* out = (float*)d_out;

    // carve workspace (256B-aligned sub-buffers)
    char* ws = (char*)d_ws;
    size_t off = 0;
    auto carve = [&](size_t bytes) -> char* {
        char* p = ws + off;
        off = (off + bytes + 255) & ~(size_t)255;
        return p;
    };
    float*    degO  = (float*)   carve((size_t)N * 4);
    float*    degI  = (float*)   carve((size_t)N * 4);
    float*    normS = (float*)   carve((size_t)N * 4);
    float*    normD = (float*)   carve((size_t)N * 4);
    float*    agg   = (float*)   carve((size_t)N * H2P * 4);   // max K = 224
    _Float16* h16   = (_Float16*)carve((size_t)N * H2P * 2);
    float*    hA    = (float*)   carve((size_t)N * H1P * 4);   // layer-1 activations
    float*    hB    = (float*)   carve((size_t)N * H2P * 4);   // layer-2 activations
    _Float16* W1p   = (_Float16*)carve((size_t)IN_F * H1P * 2);
    _Float16* W2p   = (_Float16*)carve((size_t)H1P * H2P * 2);
    _Float16* W3p   = (_Float16*)carve((size_t)H2P * OUTF * 2);
    float*    b1p   = (float*)   carve(H1P * 4);
    float*    b2p   = (float*)   carve(H2P * 4);
    float*    b3p   = (float*)   carve(OUTF * 4);

    // ---- degrees & norms
    hipMemsetAsync(degO, 0, (size_t)N * 4, stream);
    hipMemsetAsync(degI, 0, (size_t)N * 4, stream);
    gcn_degree_kernel<<<1024, 256, 0, stream>>>(es, ed, degO, degI, E);
    gcn_norm_kernel<<<(N + 255) / 256, 256, 0, stream>>>(degO, degI, normS, normD, N);

    // ---- pack weights / biases (padded, WMMA B register image)
    gcn_packw_kernel<<<(IN_F * H1P + 255) / 256, 256, 0, stream>>>(W1, W1p, IN_F, H1, IN_F, H1P);
    gcn_packw_kernel<<<(H1P * H2P + 255) / 256, 256, 0, stream>>>(W2, W2p, H1, H2, H1P, H2P);
    gcn_packw_kernel<<<(H2P * OUTF + 255) / 256, 256, 0, stream>>>(W3, W3p, H2, OUTF, H2P, OUTF);
    gcn_padbias_kernel<<<1, 256, 0, stream>>>(b1, b1p, H1, H1P);
    gcn_padbias_kernel<<<1, 256, 0, stream>>>(b2, b2p, H2, H2P);
    gcn_padbias_kernel<<<1, 256, 0, stream>>>(b3, b3p, OUTF, OUTF);

    const int MT = N / 16;  // 6250 row tiles

    // ---- layer 1: x(128) -> elu -> hA(128 padded, 100 real)
    hipMemsetAsync(agg, 0, (size_t)N * IN_F * 4, stream);
    gcn_scatter_kernel<<<4096, 256, 0, stream>>>(x, IN_F, es, ed, normS, agg, IN_F, IN_F, E);
    gcn_convert_kernel<<<((size_t)N * IN_F + 255) / 256, 256, 0, stream>>>(agg, normD, h16, IN_F, N * IN_F);
    {
        int numNT = H1P / 16, KT = IN_F / 32, tiles = MT * numNT;
        gcn_gemm_wmma_kernel<<<(tiles + 7) / 8, 256, 0, stream>>>(
            h16, IN_F, W1p, b1p, hA, H1P, numNT, KT, /*elu*/0, tiles);
    }

    // ---- layer 2: hA(100 real/128 pad) -> elu -> hB(224 padded, 200 real)
    hipMemsetAsync(agg, 0, (size_t)N * H1P * 4, stream);
    gcn_scatter_kernel<<<4096, 256, 0, stream>>>(hA, H1P, es, ed, normS, agg, H1P, H1, E);
    gcn_convert_kernel<<<((size_t)N * H1P + 255) / 256, 256, 0, stream>>>(agg, normD, h16, H1P, N * H1P);
    {
        int numNT = H2P / 16, KT = H1P / 32, tiles = MT * numNT;
        gcn_gemm_wmma_kernel<<<(tiles + 7) / 8, 256, 0, stream>>>(
            h16, H1P, W2p, b2p, hB, H2P, numNT, KT, /*elu*/0, tiles);
    }

    // ---- layer 3: hB(200 real/224 pad) -> sigmoid -> d_out(64)
    hipMemsetAsync(agg, 0, (size_t)N * H2P * 4, stream);
    gcn_scatter_kernel<<<4096, 256, 0, stream>>>(hB, H2P, es, ed, normS, agg, H2P, H2, E);
    gcn_convert_kernel<<<((size_t)N * H2P + 255) / 256, 256, 0, stream>>>(agg, normD, h16, H2P, N * H2P);
    {
        int numNT = OUTF / 16, KT = H2P / 32, tiles = MT * numNT;
        gcn_gemm_wmma_kernel<<<(tiles + 7) / 8, 256, 0, stream>>>(
            h16, H2P, W3p, b3p, out, OUTF, numNT, KT, /*sigmoid*/1, tiles);
    }
}
